// MILE_63385127355036
// MI455X (gfx1250) — matrix operand order repaired and checked by reading, not compiled, not run on hardware
//
#include <hip/hip_runtime.h>
#include <hip/hip_bf16.h>
#include <math.h>

// ---------------------------------------------------------------------------
// Problem constants (from the reference)
// ---------------------------------------------------------------------------
#define BB   512
#define TT   64
#define SS   256
#define HH   1024
#define ED   512
#define EE   1024
#define VV   2048
#define SUM  2048

typedef __attribute__((ext_vector_type(16))) __bf16 v16bf;
typedef __attribute__((ext_vector_type(8)))  float  v8f;

__device__ __forceinline__ unsigned short f2bf(float f) {
    unsigned int u = __float_as_uint(f);
    u += 0x7FFFu + ((u >> 16) & 1u);          // round-to-nearest-even
    return (unsigned short)(u >> 16);
}
__device__ __forceinline__ float sigmoidf_(float x) {
    return 1.0f / (1.0f + __expf(-x));
}

// ---------------------------------------------------------------------------
// Weight packing: fp32 [N x K] (optionally concat of two K-blocks, Wih|Whh)
// -> bf16 WMMA B-tile layout:
//   packed[((nt*KT + kt)*32 + lane)*16 + j]
//   n = nt*16 + lane%16 ; k = kt*32 + (lane/16)*16 + j
// Each lane's 16 bf16 for a tile are contiguous (one 32B load).
// ---------------------------------------------------------------------------
__global__ void pack_w_kernel(const float* __restrict__ W1, int K1,
                              const float* __restrict__ W2, int K2,
                              int N, unsigned short* __restrict__ out)
{
    const int Ktot = K1 + K2;
    const int KT = Ktot >> 5;
    const int NT = N >> 4;
    long long id = (long long)blockIdx.x * 256 + threadIdx.x;
    long long total = (long long)NT * KT * 32;
    if (id >= total) return;
    const int lane = (int)(id & 31);
    long long tk = id >> 5;
    const int kt = (int)(tk % KT);
    const int nt = (int)(tk / KT);
    const int n = (nt << 4) + (lane & 15);
    const int kstart = (kt << 5) + ((lane >> 4) << 4);
    unsigned short* o = out + id * 16;
#pragma unroll
    for (int j = 0; j < 16; ++j) {
        const int k = kstart + j;
        const float v = (k < K1) ? W1[(long long)n * K1 + k]
                                 : W2[(long long)n * K2 + (k - K1)];
        o[j] = f2bf(v);
    }
}

// ---------------------------------------------------------------------------
// Activation packing: fp32 [BB x (K1+K2)] (concat of two sources, optional
// row-gather on source 1 for embedding lookup) -> bf16 WMMA A-tile layout:
//   packed[((mt*KT + kt)*32 + lane)*16 + j]
//   m = mt*16 + lane%16
//   k = kt*32 + (lane/16)*16 + j
// (contiguous k order, same as the B packing)
// ---------------------------------------------------------------------------
__global__ void pack_a_kernel(const float* __restrict__ X1, int K1,
                              const int* __restrict__ tok, int tokStride,
                              const float* __restrict__ X2, int K2,
                              unsigned short* __restrict__ out)
{
    const int KT = (K1 + K2) >> 5;
    long long id = (long long)blockIdx.x * 256 + threadIdx.x;
    long long total = (long long)(BB / 16) * KT * 32;
    if (id >= total) return;
    const int lane = (int)(id & 31);
    long long tk = id >> 5;
    const int kt = (int)(tk % KT);
    const int mt = (int)(tk / KT);
    const int m = (mt << 4) + (lane & 15);
    const int row1 = tok ? tok[(long long)m * tokStride] : m;
    const int kb = (kt << 5) + ((lane < 16) ? 0 : 16);
    unsigned short* o = out + id * 16;
#pragma unroll
    for (int j = 0; j < 16; ++j) {
        const int k = kb + j;
        const float v = (k < K1) ? X1[(long long)row1 * K1 + k]
                                 : X2[(long long)m * K2 + (k - K1)];
        o[j] = f2bf(v);
    }
}

// ---------------------------------------------------------------------------
// Generic GEMM:  Y[BB x N] = act( A @ Wp^T + bias ),  A pre-packed bf16.
// Each wave computes a 16(M) x 32(N) strip = 2 accumulators; A loaded once
// per K-step and reused for both WMMAs. 8 waves / block.
// ---------------------------------------------------------------------------
__global__ void wmma_gemm_act(const unsigned short* __restrict__ Ap,
                              const unsigned short* __restrict__ Bp,
                              const float* __restrict__ bias,
                              float* __restrict__ Y, long long ldY,
                              int N, int KT, int act)
{
    const int tid  = threadIdx.x;
    const int lane = tid & 31;
    const int wave = tid >> 5;
    const int NT2 = N >> 5;                       // pairs of 16-wide N tiles
    const int tile = blockIdx.x * 8 + wave;
    const int mt = tile / NT2, np = tile % NT2;
    const int m0 = mt << 4;

    const v16bf* A  = (const v16bf*)Ap + (long long)mt * KT * 32 + lane;
    const v16bf* B0 = (const v16bf*)Bp + (long long)(np * 2 + 0) * KT * 32 + lane;
    const v16bf* B1 = (const v16bf*)Bp + (long long)(np * 2 + 1) * KT * 32 + lane;

    v8f acc0 = {}, acc1 = {};
#pragma unroll 2
    for (int kt = 0; kt < KT; ++kt) {
        const v16bf a  = A[(long long)kt * 32];
        const v16bf b0 = B0[(long long)kt * 32];
        const v16bf b1 = B1[(long long)kt * 32];
        acc0 = __builtin_amdgcn_wmma_f32_16x16x32_bf16(false, a, false, b0, (short)0, acc0, false, false);
        acc1 = __builtin_amdgcn_wmma_f32_16x16x32_bf16(false, a, false, b1, (short)0, acc1, false, false);
    }
    const int nc = lane & 15;
    const int n0 = (np << 5) + nc;
    const int n1 = n0 + 16;
    const float bv0 = bias ? bias[n0] : 0.0f;
    const float bv1 = bias ? bias[n1] : 0.0f;
#pragma unroll
    for (int r = 0; r < 8; ++r) {
        const int m = m0 + r + ((lane >> 4) << 3);
        float v0 = acc0[r] + bv0;
        float v1 = acc1[r] + bv1;
        if (act == 1) { v0 = tanhf(v0); v1 = tanhf(v1); }
        Y[(long long)m * ldY + n0] = v0;
        Y[(long long)m * ldY + n1] = v1;
    }
}

// ---------------------------------------------------------------------------
// Fused LSTM layer: gates = [X|Hprev] @ [Wih|Whh]^T + bih + bhh ; cell update.
// A pre-packed bf16 ([X|Hprev] concat). Each wave owns one 16x16 tile of H ->
// 4 gate accumulators (i,f,g,o at column offsets 0/H/2H/3H of packed N=4H).
// ---------------------------------------------------------------------------
__global__ void wmma_lstm(const unsigned short* __restrict__ Ap, int KT,
                          const float* __restrict__ Cprev,
                          const unsigned short* __restrict__ Wp,
                          const float* __restrict__ bih,
                          const float* __restrict__ bhh,
                          float* __restrict__ Hnew,
                          float* __restrict__ Cnew)
{
    const int tid  = threadIdx.x;
    const int lane = tid & 31;
    const int wave = tid >> 5;
    const int NT = HH >> 4;                       // 64 tiles across H
    const int tile = blockIdx.x * 8 + wave;
    const int mt = tile / NT, nt = tile % NT;
    const int m0 = mt << 4;

    const v16bf* A  = (const v16bf*)Ap + (long long)mt * KT * 32 + lane;
    const v16bf* BI = (const v16bf*)Wp + (long long)(0 * NT + nt) * KT * 32 + lane;
    const v16bf* BF = (const v16bf*)Wp + (long long)(1 * NT + nt) * KT * 32 + lane;
    const v16bf* BG = (const v16bf*)Wp + (long long)(2 * NT + nt) * KT * 32 + lane;
    const v16bf* BO = (const v16bf*)Wp + (long long)(3 * NT + nt) * KT * 32 + lane;

    v8f aI = {}, aF = {}, aG = {}, aO = {};
    for (int kt = 0; kt < KT; ++kt) {
        const v16bf a  = A[(long long)kt * 32];
        const v16bf bI = BI[(long long)kt * 32];
        const v16bf bF = BF[(long long)kt * 32];
        const v16bf bG = BG[(long long)kt * 32];
        const v16bf bO = BO[(long long)kt * 32];
        aI = __builtin_amdgcn_wmma_f32_16x16x32_bf16(false, a, false, bI, (short)0, aI, false, false);
        aF = __builtin_amdgcn_wmma_f32_16x16x32_bf16(false, a, false, bF, (short)0, aF, false, false);
        aG = __builtin_amdgcn_wmma_f32_16x16x32_bf16(false, a, false, bG, (short)0, aG, false, false);
        aO = __builtin_amdgcn_wmma_f32_16x16x32_bf16(false, a, false, bO, (short)0, aO, false, false);
    }
    const int n = (nt << 4) + (lane & 15);
    const float bI = bih[0 * HH + n] + bhh[0 * HH + n];
    const float bF = bih[1 * HH + n] + bhh[1 * HH + n];
    const float bG = bih[2 * HH + n] + bhh[2 * HH + n];
    const float bO = bih[3 * HH + n] + bhh[3 * HH + n];
#pragma unroll
    for (int r = 0; r < 8; ++r) {
        const int m = m0 + r + ((lane >> 4) << 3);
        const long long off = (long long)m * HH + n;
        const float gi = sigmoidf_(aI[r] + bI);
        const float gf = sigmoidf_(aF[r] + bF);
        const float gg = tanhf(aG[r] + bG);
        const float go = sigmoidf_(aO[r] + bO);
        const float c  = gf * Cprev[off] + gi * gg;
        Cnew[off] = c;
        Hnew[off] = go * tanhf(c);
    }
}

// ---------------------------------------------------------------------------
// Attention: per batch row, scores = enc[b] @ q[b] ; softmax ; cross = att@enc
// Block = 256 threads handles one row. Coalesced k-across-lanes dot products.
// ---------------------------------------------------------------------------
__global__ void attention_kernel(const float* __restrict__ q,
                                 const float* __restrict__ enc,
                                 float* __restrict__ cross)
{
    const int b = blockIdx.x;
    __shared__ float sQ[EE];
    __shared__ float sS[SS];
    __shared__ float red[8];
    const int tid = threadIdx.x, lane = tid & 31, wave = tid >> 5;

    for (int i = tid; i < EE; i += 256) sQ[i] = q[(long long)b * EE + i];
    __syncthreads();

    const float* eb = enc + (long long)b * SS * EE;
    for (int s = wave; s < SS; s += 8) {
        const float* er = eb + (long long)s * EE;
        float p = 0.0f;
#pragma unroll 4
        for (int j = 0; j < EE / 32; ++j) p += sQ[j * 32 + lane] * er[j * 32 + lane];
#pragma unroll
        for (int off = 16; off >= 1; off >>= 1) p += __shfl_xor(p, off, 32);
        if (lane == 0) sS[s] = p;
    }
    __syncthreads();

    // softmax over 256 scores
    const float v = sS[tid];
    float m = v;
#pragma unroll
    for (int off = 16; off >= 1; off >>= 1) m = fmaxf(m, __shfl_xor(m, off, 32));
    if (lane == 0) red[wave] = m;
    __syncthreads();
    float gm = red[0];
#pragma unroll
    for (int w = 1; w < 8; ++w) gm = fmaxf(gm, red[w]);
    const float e = __expf(v - gm);
    float ssum = e;
#pragma unroll
    for (int off = 16; off >= 1; off >>= 1) ssum += __shfl_xor(ssum, off, 32);
    __syncthreads();
    if (lane == 0) red[wave] = ssum;
    __syncthreads();
    float gs = 0.0f;
#pragma unroll
    for (int w = 0; w < 8; ++w) gs += red[w];
    sS[tid] = e / gs;
    __syncthreads();

    // cross[b,k] = sum_s att[s] * enc[b,s,k]   (thread owns 4 k's)
    float a0 = 0.f, a1 = 0.f, a2 = 0.f, a3 = 0.f;
    for (int s = 0; s < SS; ++s) {
        const float a = sS[s];
        const float* er = eb + (long long)s * EE;
        a0 += a * er[tid];
        a1 += a * er[tid + 256];
        a2 += a * er[tid + 512];
        a3 += a * er[tid + 768];
    }
    float* cr = cross + (long long)b * EE;
    cr[tid]       = a0;
    cr[tid + 256] = a1;
    cr[tid + 512] = a2;
    cr[tid + 768] = a3;
}

// ---------------------------------------------------------------------------
// decoder_init scatter: tanh(summary@W_init^T) reshaped (B,2,2,H) -> h0,h1,c0,c1
// ---------------------------------------------------------------------------
__global__ void scatter_init(const float* __restrict__ initOut,
                             float* __restrict__ h0, float* __restrict__ h1,
                             float* __restrict__ c0, float* __restrict__ c1)
{
    long long id = (long long)blockIdx.x * 256 + threadIdx.x;   // [0, B*4H)
    if (id >= (long long)BB * 4 * HH) return;
    const int n = (int)(id & (4 * HH - 1));
    const long long mb = id >> 12;
    const float v = initOut[id];
    const int which = n >> 10;          // 0:h0 1:h1 2:c0 3:c1
    const int col = n & (HH - 1);
    float* dst = (which == 0) ? h0 : (which == 1) ? h1 : (which == 2) ? c0 : c1;
    dst[mb * HH + col] = v;
}

__global__ void zero_f32(float* __restrict__ p, long long n)
{
    long long i = (long long)blockIdx.x * 256 + threadIdx.x;
    if (i < n) p[i] = 0.0f;
}

// ---------------------------------------------------------------------------
// Row argmax over V=2048 logits; first-max tie-break (min index).
// ---------------------------------------------------------------------------
__global__ void argmax_kernel(const float* __restrict__ logitsBase, long long ldRow,
                              float* __restrict__ tokBase, int t)
{
    const int b = blockIdx.x;
    const float* row = logitsBase + (long long)b * ldRow;
    __shared__ float smax[256];
    __shared__ int   sidx[256];
    float best = -INFINITY; int bi = 0;
    for (int i = threadIdx.x; i < VV; i += 256) {
        const float v = row[i];
        if (v > best) { best = v; bi = i; }
    }
    smax[threadIdx.x] = best; sidx[threadIdx.x] = bi;
    __syncthreads();
    for (int s = 128; s > 0; s >>= 1) {
        if (threadIdx.x < s) {
            const float ov = smax[threadIdx.x + s];
            const int   oi = sidx[threadIdx.x + s];
            if (ov > smax[threadIdx.x] ||
                (ov == smax[threadIdx.x] && oi < sidx[threadIdx.x])) {
                smax[threadIdx.x] = ov; sidx[threadIdx.x] = oi;
            }
        }
        __syncthreads();
    }
    if (threadIdx.x == 0) tokBase[(long long)b * TT + t] = (float)sidx[0];
}

// ---------------------------------------------------------------------------
// Host orchestration
// ---------------------------------------------------------------------------
extern "C" void kernel_launch(void* const* d_in, const int* in_sizes, int n_in,
                              void* d_out, int out_size, void* d_ws, size_t ws_size,
                              hipStream_t stream)
{
    (void)in_sizes; (void)n_in; (void)out_size; (void)ws_size;

    const int*   y        = (const int*)  d_in[0];
    const float* enc      = (const float*)d_in[1];
    const float* summary  = (const float*)d_in[2];
    const float* embTab   = (const float*)d_in[3];
    const float* W_init   = (const float*)d_in[4];
    const float* b_init   = (const float*)d_in[5];
    const float* W_ci     = (const float*)d_in[6];
    const float* b_ci     = (const float*)d_in[7];
    const float* Wih0     = (const float*)d_in[8];
    const float* Whh0     = (const float*)d_in[9];
    const float* bih0     = (const float*)d_in[10];
    const float* bhh0     = (const float*)d_in[11];
    const float* Wih1     = (const float*)d_in[12];
    const float* Whh1     = (const float*)d_in[13];
    const float* bih1     = (const float*)d_in[14];
    const float* bhh1     = (const float*)d_in[15];
    const float* W_q      = (const float*)d_in[16];
    const float* b_q      = (const float*)d_in[17];
    const float* W_co     = (const float*)d_in[18];
    const float* b_co     = (const float*)d_in[19];
    const float* W_out    = (const float*)d_in[20];
    const float* b_out    = (const float*)d_in[21];

    float* out = (float*)d_out;                       // [B,T,V] logits then [B,T] tokens
    float* tokBase = out + (long long)BB * TT * VV;

    // -------- carve workspace --------
    unsigned char* w = (unsigned char*)d_ws;
    size_t off = 0;
    auto carve = [&](size_t bytes) -> void* {
        off = (off + 255) & ~(size_t)255;
        void* p = w + off;
        off += bytes;
        return p;
    };
    unsigned short* Wci_p   = (unsigned short*)carve((size_t)ED   * (ED + EE) * 2);
    unsigned short* Wl0_p   = (unsigned short*)carve((size_t)4*HH * (ED + HH) * 2);
    unsigned short* Wl1_p   = (unsigned short*)carve((size_t)4*HH * (HH + HH) * 2);
    unsigned short* Wq_p    = (unsigned short*)carve((size_t)EE   * (2 * HH)  * 2);
    unsigned short* Wco_p   = (unsigned short*)carve((size_t)HH   * (EE + HH) * 2);
    unsigned short* Wout_p  = (unsigned short*)carve((size_t)VV   * HH        * 2);
    unsigned short* Winit_p = (unsigned short*)carve((size_t)4*HH * SUM       * 2);
    unsigned short* Apack   = (unsigned short*)carve((size_t)BB * (2 * HH) * 2); // reused per GEMM
    float* initOut = (float*)carve((size_t)BB * 4 * HH * 4);
    float* h0[2], *h1[2], *c0[2], *c1[2], *cross[2];
    for (int i = 0; i < 2; ++i) {
        h0[i]    = (float*)carve((size_t)BB * HH * 4);
        h1[i]    = (float*)carve((size_t)BB * HH * 4);
        c0[i]    = (float*)carve((size_t)BB * HH * 4);
        c1[i]    = (float*)carve((size_t)BB * HH * 4);
        cross[i] = (float*)carve((size_t)BB * EE * 4);
    }
    float* x_ci = (float*)carve((size_t)BB * ED * 4);
    float* qb   = (float*)carve((size_t)BB * EE * 4);
    float* lin  = (float*)carve((size_t)BB * HH * 4);

    // -------- helpers --------
    auto packWBlocks = [](int N, int Ktot) {
        long long total = (long long)(N >> 4) * (Ktot >> 5) * 32;
        return (int)((total + 255) / 256);
    };
    auto packABlocks = [](int Ktot) {
        long long total = (long long)(BB / 16) * (Ktot >> 5) * 32;
        return (int)((total + 255) / 256);
    };
    auto packA = [&](const float* X1, int K1, const int* tok, int tokStride,
                     const float* X2, int K2) {
        hipLaunchKernelGGL(pack_a_kernel, dim3(packABlocks(K1 + K2)), dim3(256), 0, stream,
                           X1, K1, tok, tokStride, X2, K2, Apack);
    };
    auto gemm = [&](const unsigned short* Bp, const float* bias, float* Y,
                    long long ldY, int N, int Ktot, int act) {
        hipLaunchKernelGGL(wmma_gemm_act, dim3(N / 8), dim3(256), 0, stream,
                           Apack, Bp, bias, Y, ldY, N, Ktot >> 5, act);
    };

    // -------- pack weights to bf16 WMMA B layout --------
    hipLaunchKernelGGL(pack_w_kernel, dim3(packWBlocks(ED, ED + EE)), dim3(256), 0, stream,
                       W_ci, ED + EE, (const float*)nullptr, 0, ED, Wci_p);
    hipLaunchKernelGGL(pack_w_kernel, dim3(packWBlocks(4 * HH, ED + HH)), dim3(256), 0, stream,
                       Wih0, ED, Whh0, HH, 4 * HH, Wl0_p);
    hipLaunchKernelGGL(pack_w_kernel, dim3(packWBlocks(4 * HH, 2 * HH)), dim3(256), 0, stream,
                       Wih1, HH, Whh1, HH, 4 * HH, Wl1_p);
    hipLaunchKernelGGL(pack_w_kernel, dim3(packWBlocks(EE, 2 * HH)), dim3(256), 0, stream,
                       W_q, 2 * HH, (const float*)nullptr, 0, EE, Wq_p);
    hipLaunchKernelGGL(pack_w_kernel, dim3(packWBlocks(HH, EE + HH)), dim3(256), 0, stream,
                       W_co, EE + HH, (const float*)nullptr, 0, HH, Wco_p);
    hipLaunchKernelGGL(pack_w_kernel, dim3(packWBlocks(VV, HH)), dim3(256), 0, stream,
                       W_out, HH, (const float*)nullptr, 0, VV, Wout_p);
    hipLaunchKernelGGL(pack_w_kernel, dim3(packWBlocks(4 * HH, SUM)), dim3(256), 0, stream,
                       W_init, SUM, (const float*)nullptr, 0, 4 * HH, Winit_p);

    // -------- decoder init: tanh(summary @ W_init^T + b) -> h/c state --------
    packA(summary, SUM, nullptr, 0, nullptr, 0);
    gemm(Winit_p, b_init, initOut, 4 * HH, 4 * HH, SUM, 1);
    hipLaunchKernelGGL(scatter_init, dim3((BB * 4 * HH) / 256), dim3(256), 0, stream,
                       initOut, h0[0], h1[0], c0[0], c1[0]);
    hipLaunchKernelGGL(zero_f32, dim3((BB * EE) / 256), dim3(256), 0, stream,
                       cross[0], (long long)BB * EE);

    // -------- the T-step scan --------
    for (int t = 0; t < TT; ++t) {
        const int rp = t & 1;
        const int wp = rp ^ 1;

        // 1) x_ci = tanh([emb_table[y[:,t]] | cross] @ W_ci^T + b_ci)
        packA(embTab, ED, y + t, TT, cross[rp], EE);
        gemm(Wci_p, b_ci, x_ci, ED, ED, ED + EE, 1);

        // 2) LSTM layer 0 (fused gates + cell)
        packA(x_ci, ED, nullptr, 0, h0[rp], HH);
        hipLaunchKernelGGL(wmma_lstm, dim3((BB / 16) * (HH / 16) / 8), dim3(256), 0, stream,
                           Apack, (ED + HH) >> 5, c0[rp], Wl0_p, bih0, bhh0,
                           h0[wp], c0[wp]);

        // 3) LSTM layer 1
        packA(h0[wp], HH, nullptr, 0, h1[rp], HH);
        hipLaunchKernelGGL(wmma_lstm, dim3((BB / 16) * (HH / 16) / 8), dim3(256), 0, stream,
                           Apack, (2 * HH) >> 5, c1[rp], Wl1_p, bih1, bhh1,
                           h1[wp], c1[wp]);

        // 4) q = tanh([h1|c1] @ W_q^T + b_q)
        packA(h1[wp], HH, nullptr, 0, c1[wp], HH);
        gemm(Wq_p, b_q, qb, EE, EE, 2 * HH, 1);

        // 5) attention -> cross_new
        hipLaunchKernelGGL(attention_kernel, dim3(BB), dim3(256), 0, stream,
                           qb, enc, cross[wp]);

        // 6) lin = tanh([h1|cross_new] @ W_co^T + b_co)
        packA(h1[wp], HH, nullptr, 0, cross[wp], EE);
        gemm(Wco_p, b_co, lin, HH, HH, EE + HH, 1);

        // 7) logits[:, t, :] = lin @ W_out^T + b_out  (straight into d_out)
        packA(lin, HH, nullptr, 0, nullptr, 0);
        gemm(Wout_p, b_out, out + (long long)t * VV, (long long)TT * VV, VV, HH, 0);

        // 8) tokens[:, t] = argmax(logits[:, t, :])
        hipLaunchKernelGGL(argmax_kernel, dim3(BB), dim3(256), 0, stream,
                           out + (long long)t * VV, (long long)TT * VV, tokBase, t);
    }
}
